// GNNLayer_8383776162378
// MI455X (gfx1250) — matrix operand order repaired and checked by reading, not compiled, not run on hardware
//
#include <hip/hip_runtime.h>
#include <hip/hip_bf16.h>
#include <math.h>

#define N_NODES 25000
#define N_EDGES 400000
#define DD      128
#define HEADS   4
#define GG      32
#define EPSV    1e-5f
#define NT      2      // n-tiles (16 cols each) per wave; block of 4 waves = 128 cols

typedef __bf16 bf16_t;
typedef bf16_t v16bf __attribute__((ext_vector_type(16)));
typedef float  v8f   __attribute__((ext_vector_type(8)));

__device__ __forceinline__ unsigned short f2bf(float f) {
    unsigned u = __float_as_uint(f);
    u += 0x7FFFu + ((u >> 16) & 1u);           // round-to-nearest-even
    return (unsigned short)(u >> 16);
}
__device__ __forceinline__ float bf2f(unsigned short b) {
    return __uint_as_float(((unsigned)b) << 16);
}
__device__ __forceinline__ void atomicMaxF(float* addr, float val) {
    unsigned* ua = (unsigned*)addr;
    unsigned cur = *ua;
    while (__uint_as_float(cur) < val) {
        unsigned assumed = cur;
        cur = atomicCAS(ua, assumed, __float_as_uint(val));
        if (cur == assumed) break;
    }
}

// ---------------------------------------------------------------- prep kernels
__global__ void k_f2bf(const float* __restrict__ in, unsigned short* __restrict__ out, size_t n) {
    size_t i = blockIdx.x * (size_t)blockDim.x + threadIdx.x;
    if (i < n) out[i] = f2bf(in[i]);
}
__global__ void k_fill(float* __restrict__ p, float v, size_t n) {
    size_t i = blockIdx.x * (size_t)blockDim.x + threadIdx.x;
    if (i < n) p[i] = v;
}
// Wt[n*K+k] = bf16(W[k*Nout+n])  (weights stored [in,out] in the reference)
__global__ void k_transpose_bf(const float* __restrict__ W, unsigned short* __restrict__ Wt,
                               int K, int Nout) {
    size_t i = blockIdx.x * (size_t)blockDim.x + threadIdx.x;
    if (i >= (size_t)K * Nout) return;
    int n = (int)(i / K), k = (int)(i % K);
    Wt[(size_t)n * K + k] = f2bf(W[(size_t)k * Nout + n]);
}
__global__ void k_pack_qkv_w(const float* __restrict__ Wq, const float* __restrict__ Wk,
                             const float* __restrict__ Wv, unsigned short* __restrict__ Wt) {
    int i = blockIdx.x * blockDim.x + threadIdx.x;
    if (i >= 384 * 128) return;
    int n = i / 128, k = i % 128;
    float v = (n < 128) ? Wq[k * 128 + n] : (n < 256) ? Wk[k * 128 + (n - 128)]
                                                      : Wv[k * 128 + (n - 256)];
    Wt[n * 128 + k] = f2bf(v);
}
__global__ void k_pack_qkv_b(const float* __restrict__ bq, const float* __restrict__ bk,
                             const float* __restrict__ bv, float* __restrict__ b) {
    int i = blockIdx.x * blockDim.x + threadIdx.x;
    if (i >= 384) return;
    b[i] = (i < 128) ? bq[i] : (i < 256) ? bk[i - 128] : bv[i - 256];
}

// ------------------------------------------------------------- WMMA bf16 GEMM
// C[M,Nout] = act(A[M,K] @ Wt^T + bias) + addsrc ; Wt stored [Nout][K] bf16.
// Each wave computes a 16 x (NT*16) strip: one A fragment, NT accumulators.
// Tail rows are CLAMPED (not predicated): A row r only feeds output row r, and
// out-of-range output rows are never stored, so duplicated loads are harmless.
union fragU { v16bf v; uint4 q[2]; };

__global__ __launch_bounds__(128)
void k_gemm_bf16(const unsigned short* __restrict__ A,
                 const unsigned short* __restrict__ Wt,
                 const float* __restrict__ bias,
                 const float* __restrict__ addsrc,
                 float* __restrict__ outF,
                 unsigned short* __restrict__ outB,
                 int M, int K, int Nout, int act) {
    const int lane  = threadIdx.x & 31;
    const int wv    = threadIdx.x >> 5;
    const int m0    = blockIdx.x * 16;
    const int n0    = (blockIdx.y * 4 + wv) * (16 * NT);
    const int l15   = lane & 15;
    const int row   = min(m0 + l15, M - 1);   // clamp instead of predicate
    const int koffA = (lane < 16) ? 0 : 8;    // 16-bit A frag layout (ISA 7.12.2)
    const int koffB = (lane < 16) ? 0 : 8;    // 16-bit B frag layout
    const unsigned short* arow = A + (size_t)row * K + koffA;
    const unsigned short* brow[NT];
#pragma unroll
    for (int t = 0; t < NT; ++t)
        brow[t] = Wt + (size_t)(n0 + t * 16 + l15) * K + koffB;
    v8f acc[NT];
#pragma unroll
    for (int t = 0; t < NT; ++t) acc[t] = (v8f){0.f,0.f,0.f,0.f,0.f,0.f,0.f,0.f};
    for (int kb = 0; kb < K; kb += 32) {
        fragU ua;
        ua.q[0] = *(const uint4*)(arow + kb);        // K pairs {0,2,4,6}+off
        ua.q[1] = *(const uint4*)(arow + kb + 16);   // K pairs {16,18,20,22}+off
#pragma unroll
        for (int t = 0; t < NT; ++t) {
            fragU ub;
            ub.q[0] = *(const uint4*)(brow[t] + kb);
            ub.q[1] = *(const uint4*)(brow[t] + kb + 16);
            acc[t] = __builtin_amdgcn_wmma_f32_16x16x32_bf16(false, ua.v, false, ub.v,
                                                             (short)0, acc[t], false, false);
        }
    }
    const int rbase = (lane < 16) ? 0 : 8;
#pragma unroll
    for (int t = 0; t < NT; ++t) {
        const int col = n0 + t * 16 + l15;
#pragma unroll
        for (int r = 0; r < 8; ++r) {
            int orow = m0 + rbase + r;
            if (orow >= M) continue;
            float val = acc[t][r];
            if (bias) val += bias[col];
            if (act == 1)      val = fmaxf(val, 0.f);
            else if (act == 2) val = 0.5f * val * (1.f + erff(val * 0.70710678118f));
            size_t o = (size_t)orow * Nout + col;
            if (addsrc) val += addsrc[o];
            if (outF) outF[o] = val;
            if (outB) outB[o] = f2bf(val);
        }
    }
}

// EdgeMLP GEMM1: rows are edges; A row = [h2[src] | h_E | h2[dst]] (K=384), GELU.
// Segment choice is uniform per 32-wide K chunk; fully unrolled -> no branches.
// One gathered A fragment feeds NT accumulators (halves gather traffic).
__global__ __launch_bounds__(128)
void k_gemm_edgecat(const unsigned short* __restrict__ h2b,
                    const unsigned short* __restrict__ hEb,
                    const int* __restrict__ ei,
                    const unsigned short* __restrict__ Wt,   // [128][384]
                    const float* __restrict__ bias,
                    unsigned short* __restrict__ outB,       // [E,128] bf16
                    int E) {
    const int lane  = threadIdx.x & 31;
    const int wv    = threadIdx.x >> 5;
    const int m0    = blockIdx.x * 16;
    const int l15   = lane & 15;
    const int row   = m0 + l15;                       // edge id (E % 16 == 0)
    const int n0    = (blockIdx.y * 4 + wv) * (16 * NT);
    const int koffA = (lane < 16) ? 0 : 8;
    const int koffB = (lane < 16) ? 0 : 8;
    const int s = ei[row];
    const int d = ei[E + row];
    const unsigned short* seg[3] = { h2b + (size_t)s   * 128 + koffA,
                                     hEb + (size_t)row * 128 + koffA,
                                     h2b + (size_t)d   * 128 + koffA };
    const unsigned short* brow[NT];
#pragma unroll
    for (int t = 0; t < NT; ++t)
        brow[t] = Wt + (size_t)(n0 + t * 16 + l15) * 384 + koffB;
    v8f acc[NT];
#pragma unroll
    for (int t = 0; t < NT; ++t) acc[t] = (v8f){0.f,0.f,0.f,0.f,0.f,0.f,0.f,0.f};
#pragma unroll
    for (int kb = 0; kb < 384; kb += 32) {
        const unsigned short* aptr = seg[kb >> 7] + (kb & 127);  // compile-time select
        fragU ua;
        ua.q[0] = *(const uint4*)(aptr);
        ua.q[1] = *(const uint4*)(aptr + 16);
#pragma unroll
        for (int t = 0; t < NT; ++t) {
            fragU ub;
            ub.q[0] = *(const uint4*)(brow[t] + kb);
            ub.q[1] = *(const uint4*)(brow[t] + kb + 16);
            acc[t] = __builtin_amdgcn_wmma_f32_16x16x32_bf16(false, ua.v, false, ub.v,
                                                             (short)0, acc[t], false, false);
        }
    }
    const int rbase = (lane < 16) ? 0 : 8;
#pragma unroll
    for (int t = 0; t < NT; ++t) {
        const int col = n0 + t * 16 + l15;
#pragma unroll
        for (int r = 0; r < 8; ++r) {
            int orow = m0 + rbase + r;
            float val = acc[t][r] + bias[col];
            val = 0.5f * val * (1.f + erff(val * 0.70710678118f));   // exact GELU
            outB[(size_t)orow * 128 + col] = f2bf(val);
        }
    }
}

// ---------------------------------------------------------------- attention
__global__ void k_attn_logits(const float* __restrict__ qkv, const unsigned short* __restrict__ ebf,
                              const int* __restrict__ ei, float* __restrict__ alpha,
                              float* __restrict__ amax, int E) {
    size_t i = blockIdx.x * (size_t)blockDim.x + threadIdx.x;
    if (i >= (size_t)E * HEADS) return;
    int e = (int)(i >> 2), h = (int)(i & 3);
    int s = ei[e], d = ei[E + e];
    const float*          q  = qkv + (size_t)d * 384 + h * 32;
    const float*          kk = qkv + (size_t)s * 384 + 128 + h * 32;
    const unsigned short* ev = ebf + (size_t)e * 128 + h * 32;
    float a = 0.f;
#pragma unroll
    for (int j = 0; j < 32; ++j) a += q[j] * (kk[j] + bf2f(ev[j]));
    a *= 0.17677669529663687f;                 // 1/sqrt(32)
    alpha[i] = a;
    atomicMaxF(&amax[d * 4 + h], a);
}
__global__ void k_attn_exp(const int* __restrict__ ei, float* __restrict__ alpha,
                           const float* __restrict__ amax, float* __restrict__ denom, int E) {
    size_t i = blockIdx.x * (size_t)blockDim.x + threadIdx.x;
    if (i >= (size_t)E * HEADS) return;
    int e = (int)(i >> 2), h = (int)(i & 3);
    int d = ei[E + e];
    float a = expf(alpha[i] - amax[d * 4 + h]);
    alpha[i] = a;
    atomicAdd(&denom[d * 4 + h], a);
}
__global__ void k_attn_aggr(const float* __restrict__ qkv, const unsigned short* __restrict__ ebf,
                            const int* __restrict__ ei, const float* __restrict__ alpha,
                            const float* __restrict__ denom, float* __restrict__ dh, int E) {
    size_t i = blockIdx.x * (size_t)blockDim.x + threadIdx.x;
    if (i >= (size_t)E * DD) return;
    int e = (int)(i >> 7), c = (int)(i & 127), h = c >> 5;
    int s = ei[e], d = ei[E + e];
    float w   = alpha[(size_t)e * 4 + h] / denom[d * 4 + h];
    float val = (qkv[(size_t)s * 384 + 256 + c] + bf2f(ebf[(size_t)e * 128 + c])) * w;
    atomicAdd(&dh[(size_t)d * 128 + c], val);
}

// ---------------------------------------------------------------- layernorm
__global__ __launch_bounds__(128)
void k_layernorm(const float* __restrict__ a, const float* __restrict__ b,
                 const float* __restrict__ g, const float* __restrict__ be,
                 float* __restrict__ outF, unsigned short* __restrict__ outB) {
    __shared__ float red[128];
    int row = blockIdx.x, t = threadIdx.x;
    size_t o = (size_t)row * 128 + t;
    float x = a[o] + (b ? b[o] : 0.f);
    red[t] = x; __syncthreads();
    for (int s = 64; s > 0; s >>= 1) { if (t < s) red[t] += red[t + s]; __syncthreads(); }
    float mean = red[0] * (1.f / 128.f); __syncthreads();
    float dx = x - mean;
    red[t] = dx * dx; __syncthreads();
    for (int s = 64; s > 0; s >>= 1) { if (t < s) red[t] += red[t + s]; __syncthreads(); }
    float var = red[0] * (1.f / 128.f);
    float y = g[t] * dx * rsqrtf(var + EPSV) + be[t];
    if (outF) outF[o] = y;
    if (outB) outB[o] = f2bf(y);
}

// ---------------------------------------------------------------- batchnorm
__global__ __launch_bounds__(128)
void k_bn_stats(const float* __restrict__ x, float* __restrict__ sum,
                float* __restrict__ sumsq, int rows, int rpb) {
    int d = threadIdx.x;
    int r0 = blockIdx.x * rpb;
    int r1 = min(r0 + rpb, rows);
    float s = 0.f, q = 0.f;
    for (int r = r0; r < r1; ++r) { float v = x[(size_t)r * 128 + d]; s += v; q += v * v; }
    atomicAdd(&sum[d], s); atomicAdd(&sumsq[d], q);
}
__global__ void k_bn_norm(float* __restrict__ x, const float* __restrict__ sum,
                          const float* __restrict__ sumsq, const float* __restrict__ g,
                          const float* __restrict__ b, size_t n, float invR) {
    size_t i = blockIdx.x * (size_t)blockDim.x + threadIdx.x;
    if (i >= n) return;
    int d = (int)(i & 127);
    float mean = sum[d] * invR;
    float var  = sumsq[d] * invR - mean * mean;
    x[i] = g[d] * (x[i] - mean) * rsqrtf(var + EPSV) + b[d];
}

// ---------------------------------------------------------------- context gate
__global__ void k_ctx_accum(const float* __restrict__ h2, const int* __restrict__ batch,
                            float* __restrict__ c, float* __restrict__ cnt, int N) {
    size_t i = blockIdx.x * (size_t)blockDim.x + threadIdx.x;
    if (i >= (size_t)N * 128) return;
    int n = (int)(i >> 7), d = (int)(i & 127);
    int g = batch[n];
    atomicAdd(&c[(size_t)g * 128 + d], h2[i]);
    if (d == 0) atomicAdd(&cnt[g], 1.f);
}
__global__ __launch_bounds__(128)
void k_ctx_mlp(const float* __restrict__ c, const float* __restrict__ cnt,
               const float* __restrict__ w1, const float* __restrict__ b1,
               const float* __restrict__ w2, const float* __restrict__ b2,
               float* __restrict__ gate) {
    __shared__ float cs[128];
    __shared__ float t1[128];
    int g = blockIdx.x, t = threadIdx.x;
    float inv = 1.f / fmaxf(cnt[g], 1.f);
    cs[t] = c[(size_t)g * 128 + t] * inv;
    __syncthreads();
    float a = b1[t];
    for (int k = 0; k < 128; ++k) a += cs[k] * w1[k * 128 + t];
    t1[t] = fmaxf(a, 0.f);
    __syncthreads();
    float o = b2[t];
    for (int k = 0; k < 128; ++k) o += t1[k] * w2[k * 128 + t];
    gate[(size_t)g * 128 + t] = 1.f / (1.f + expf(-o));
}
__global__ void k_gate_apply(const float* __restrict__ h2, const int* __restrict__ batch,
                             const float* __restrict__ gate, float* __restrict__ out, int N) {
    size_t i = blockIdx.x * (size_t)blockDim.x + threadIdx.x;
    if (i >= (size_t)N * 128) return;
    int n = (int)(i >> 7), d = (int)(i & 127);
    out[i] = h2[i] * gate[(size_t)batch[n] * 128 + d];
}

// ================================================================= launcher
extern "C" void kernel_launch(void* const* d_in, const int* in_sizes, int n_in,
                              void* d_out, int out_size, void* d_ws, size_t ws_size,
                              hipStream_t stream) {
    const float* h_V  = (const float*)d_in[0];
    const float* h_E  = (const float*)d_in[1];
    const int*   ei   = (const int*)d_in[2];
    const int*   bid  = (const int*)d_in[3];
    const float* Wq   = (const float*)d_in[4];  const float* bq = (const float*)d_in[5];
    const float* Wk   = (const float*)d_in[6];  const float* bk = (const float*)d_in[7];
    const float* Wv   = (const float*)d_in[8];  const float* bv = (const float*)d_in[9];
    const float* We   = (const float*)d_in[10];
    const float* ln1g = (const float*)d_in[11]; const float* ln1b = (const float*)d_in[12];
    const float* ln2g = (const float*)d_in[13]; const float* ln2b = (const float*)d_in[14];
    const float* fw1  = (const float*)d_in[15]; const float* fb1 = (const float*)d_in[16];
    const float* fw2  = (const float*)d_in[17]; const float* fb2 = (const float*)d_in[18];
    const float* W11  = (const float*)d_in[19]; const float* b11 = (const float*)d_in[20];
    const float* W12  = (const float*)d_in[21]; const float* b12 = (const float*)d_in[22];
    const float* bng  = (const float*)d_in[23]; const float* bnb = (const float*)d_in[24];
    const float* cw1  = (const float*)d_in[25]; const float* cb1 = (const float*)d_in[26];
    const float* cw2  = (const float*)d_in[27]; const float* cb2 = (const float*)d_in[28];

    const int N = N_NODES, E = N_EDGES;
    char* wsp = (char*)d_ws;
    size_t off = 0;
    auto alloc = [&](size_t bytes) -> void* {
        void* p = wsp + off; off = (off + bytes + 255) & ~(size_t)255; return p;
    };
    unsigned short* hVb   = (unsigned short*)alloc((size_t)N * 128 * 2);
    unsigned short* hEb   = (unsigned short*)alloc((size_t)E * 128 * 2);
    float*          qkv   = (float*)alloc((size_t)N * 384 * 4);
    unsigned short* ebf   = (unsigned short*)alloc((size_t)E * 128 * 2); // reused as m1
    float*          alpha = (float*)alloc((size_t)E * 4 * 4);
    float*          amax  = (float*)alloc((size_t)N * 4 * 4);
    float*          denom = (float*)alloc((size_t)N * 4 * 4);
    float*          dh    = (float*)alloc((size_t)N * 128 * 4);          // reused as y
    float*          h1F   = (float*)alloc((size_t)N * 128 * 4);
    unsigned short* h1B   = (unsigned short*)alloc((size_t)N * 128 * 2);
    unsigned short* midB  = (unsigned short*)alloc((size_t)N * 512 * 2);
    float*          h2F   = (float*)alloc((size_t)N * 128 * 4);
    unsigned short* h2B   = (unsigned short*)alloc((size_t)N * 128 * 2);
    unsigned short* WqkvT = (unsigned short*)alloc(384 * 128 * 2);
    float*          bqkv  = (float*)alloc(384 * 4);
    unsigned short* WeT   = (unsigned short*)alloc(128 * 128 * 2);
    unsigned short* W1T   = (unsigned short*)alloc(512 * 128 * 2);
    unsigned short* W2T   = (unsigned short*)alloc(128 * 512 * 2);
    unsigned short* W11T  = (unsigned short*)alloc(128 * 384 * 2);
    unsigned short* W12T  = (unsigned short*)alloc(128 * 128 * 2);
    float*          bnS   = (float*)alloc(128 * 4);
    float*          bnQ   = (float*)alloc(128 * 4);
    float*          ctxC  = (float*)alloc(GG * 128 * 4);
    float*          ctxN  = (float*)alloc(GG * 4);
    float*          gate  = (float*)alloc(GG * 128 * 4);
    (void)ws_size; (void)n_in; (void)in_sizes; (void)out_size;

    float* out_h  = (float*)d_out;                   // [N,128]
    float* out_hE = (float*)d_out + (size_t)N * 128; // [E,128]

    const int T = 256;
    auto blks = [](size_t n, int t) { return (unsigned)((n + t - 1) / t); };
    const int MT_N = (N + 15) / 16;   // 1563 node-row tiles
    const int MT_E = E / 16;          // 25000 edge-row tiles
    // with NT=2, a 4-wave block covers 128 output columns
    const int COLS_PER_BLK = 4 * 16 * NT;   // 128

    // ---- prep: bf16 conversions + weight transposes -------------------------
    k_f2bf<<<blks((size_t)N * 128, T), T, 0, stream>>>(h_V, hVb, (size_t)N * 128);
    k_f2bf<<<blks((size_t)E * 128, T), T, 0, stream>>>(h_E, hEb, (size_t)E * 128);
    k_pack_qkv_w<<<blks(384 * 128, T), T, 0, stream>>>(Wq, Wk, Wv, WqkvT);
    k_pack_qkv_b<<<2, T, 0, stream>>>(bq, bk, bv, bqkv);
    k_transpose_bf<<<blks(128 * 128, T), T, 0, stream>>>(We,  WeT,  128, 128);
    k_transpose_bf<<<blks(128 * 512, T), T, 0, stream>>>(fw1, W1T,  128, 512);
    k_transpose_bf<<<blks(512 * 128, T), T, 0, stream>>>(fw2, W2T,  512, 128);
    k_transpose_bf<<<blks(384 * 128, T), T, 0, stream>>>(W11, W11T, 384, 128);
    k_transpose_bf<<<blks(128 * 128, T), T, 0, stream>>>(W12, W12T, 128, 128);
    k_fill<<<blks((size_t)N * 4, T), T, 0, stream>>>(amax, -3.0e38f, (size_t)N * 4);
    k_fill<<<blks((size_t)N * 4, T), T, 0, stream>>>(denom, 0.f, (size_t)N * 4);
    k_fill<<<blks((size_t)N * 128, T), T, 0, stream>>>(dh, 0.f, (size_t)N * 128);
    k_fill<<<1, T, 0, stream>>>(bnS, 0.f, 128);
    k_fill<<<1, T, 0, stream>>>(bnQ, 0.f, 128);
    k_fill<<<blks(GG * 128, T), T, 0, stream>>>(ctxC, 0.f, GG * 128);
    k_fill<<<1, T, 0, stream>>>(ctxN, 0.f, GG);

    // ---- TransformerConv -----------------------------------------------------
    k_gemm_bf16<<<dim3(MT_N, 384 / COLS_PER_BLK), 128, 0, stream>>>(
        hVb, WqkvT, bqkv, nullptr, qkv, nullptr, N, 128, 384, 0);
    k_gemm_bf16<<<dim3(MT_E, 128 / COLS_PER_BLK), 128, 0, stream>>>(
        hEb, WeT, nullptr, nullptr, nullptr, ebf, E, 128, 128, 0);
    k_attn_logits<<<blks((size_t)E * 4, T), T, 0, stream>>>(qkv, ebf, ei, alpha, amax, E);
    k_attn_exp<<<blks((size_t)E * 4, T), T, 0, stream>>>(ei, alpha, amax, denom, E);
    k_attn_aggr<<<blks((size_t)E * 128, T), T, 0, stream>>>(qkv, ebf, ei, alpha, denom, dh, E);
    k_layernorm<<<N, 128, 0, stream>>>(h_V, dh, ln1g, ln1b, h1F, h1B);

    // ---- FFN + LN2 -----------------------------------------------------------
    k_gemm_bf16<<<dim3(MT_N, 512 / COLS_PER_BLK), 128, 0, stream>>>(
        h1B, W1T, fb1, nullptr, nullptr, midB, N, 128, 512, 1);
    k_gemm_bf16<<<dim3(MT_N, 128 / COLS_PER_BLK), 128, 0, stream>>>(
        midB, W2T, fb2, h1F, dh, nullptr, N, 512, 128, 0);
    k_layernorm<<<N, 128, 0, stream>>>(dh, nullptr, ln2g, ln2b, h2F, h2B);

    // ---- EdgeMLP + BatchNorm -------------------------------------------------
    k_gemm_edgecat<<<dim3(MT_E, 128 / COLS_PER_BLK), 128, 0, stream>>>(
        h2B, hEb, ei, W11T, b11, ebf, E);
    k_gemm_bf16<<<dim3(MT_E, 128 / COLS_PER_BLK), 128, 0, stream>>>(
        ebf, W12T, b12, h_E, out_hE, nullptr, E, 128, 128, 0);
    const int RPB = 1024;
    k_bn_stats<<<blks(E, RPB), 128, 0, stream>>>(out_hE, bnS, bnQ, E, RPB);
    k_bn_norm<<<blks((size_t)E * 128, T), T, 0, stream>>>(out_hE, bnS, bnQ, bng, bnb,
                                                          (size_t)E * 128, 1.f / (float)E);

    // ---- Context gating ------------------------------------------------------
    k_ctx_accum<<<blks((size_t)N * 128, T), T, 0, stream>>>(h2F, bid, ctxC, ctxN, N);
    k_ctx_mlp<<<GG, 128, 0, stream>>>(ctxC, ctxN, cw1, cb1, cw2, cb2, gate);
    k_gate_apply<<<blks((size_t)N * 128, T), T, 0, stream>>>(h2F, bid, gate, out_h, N);
}
